// PureEmergencePDE_59725815218186
// MI455X (gfx1250) — matrix operand order repaired and checked by reading, not compiled, not run on hardware
//
#include <hip/hip_runtime.h>
#include <hip/hip_bf16.h>
#include <math.h>

// ---------------------------------------------------------------------------
// PureEmergencePDE on MI455X (gfx1250): WMMA f16 GEMMs for the per-cell MLP,
// fused with the multi-dilation Laplacian stencil, gated memory add, Euler
// update and periodic RMS norm. wave32, 16x16x32 f16 WMMA.
// ---------------------------------------------------------------------------

#define CCH   128          // channels C
#define HID   256          // hidden = 2C
#define HDIM  128
#define WDIM  128
#define BDIM  8
#define HWSZ  (HDIM * WDIM)
#define TCELL 64           // cells per workgroup (half a row)

typedef _Float16 f16;
typedef _Float16 v16h __attribute__((ext_vector_type(16)));
typedef float    v8f  __attribute__((ext_vector_type(8)));

union Frag { uint4 q[2]; v16h v; };

__device__ __forceinline__ float sigmoidf_(float x) {
    return 1.0f / (1.0f + __expf(-x));
}

// A-fragment (16x32 f16) from LDS, row-major with `stride` halves per row.
// Lane L holds row (L&15); half-select hs=(L>>4) picks K = {8*hs..8*hs+7} and
// {16+8*hs..16+8*hs+7} (two contiguous 16B runs) per CDNA5 ISA 7.12.2.
__device__ __forceinline__ v16h load_a_frag(const f16* __restrict__ lds,
                                            int stride, int rowBase, int kb,
                                            int lane) {
    const int r  = rowBase + (lane & 15);
    const int hs = lane >> 4;
    const f16* p = lds + r * stride + kb + 8 * hs;
    Frag f;
    f.q[0] = *(const uint4*)(p);
    f.q[1] = *(const uint4*)(p + 16);
    return f.v;
}

// B-fragment (32x16 f16) from a K-contiguous (transposed) weight matrix in
// global memory: lane L holds column (L&15); hs=(L>>4) picks K = {8*hs..8*hs+7}
// and {16+8*hs..16+8*hs+7} (two contiguous 16B runs), matching load_a_frag.
__device__ __forceinline__ v16h load_b_frag(const f16* __restrict__ g,
                                            int stride, int colBase, int kb,
                                            int lane) {
    const int c  = colBase + (lane & 15);
    const int hs = lane >> 4;
    const f16* p = g + c * stride + kb + 8 * hs;
    Frag f;
    f.q[0] = *(const uint4*)(p);
    f.q[1] = *(const uint4*)(p + 16);
    return f.v;
}

__device__ __forceinline__ int reflect_(int i, int n) {
    if (i < 0) i = -i;
    if (i >= n) i = 2 * n - 2 - i;
    return i;
}

// ---------------------------------------------------------------------------
// setup: W1 (C x 2C) -> W1t f16 (2C x C, K contiguous);
//        W2 (2C x C) -> W2t f16 (C x 2C, K contiguous); zero h.
// ---------------------------------------------------------------------------
__global__ void pde_setup_kernel(const float* __restrict__ W1,
                                 const float* __restrict__ W2,
                                 f16* __restrict__ W1t, f16* __restrict__ W2t,
                                 float* __restrict__ h) {
    int idx = blockIdx.x * 256 + threadIdx.x;
    if (idx < HID * CCH) {                      // 32768: W1t[n*128+k]
        int n = idx >> 7, k = idx & 127;
        W1t[idx] = (f16)W1[k * HID + n];
    } else if (idx < 2 * HID * CCH) {           // W2t[n*256+k]
        int j = idx - HID * CCH;
        int n = j >> 8, k = j & 255;
        W2t[j] = (f16)W2[k * CCH + n];
    } else if (idx < 2 * HID * CCH + BDIM * CCH) {
        h[idx - 2 * HID * CCH] = 0.0f;
    }
}

// ---------------------------------------------------------------------------
// per-(b,c) plane reduction: pooled mean (up) and mean |u| (am)
// ---------------------------------------------------------------------------
__global__ __launch_bounds__(256)
void pde_reduce_kernel(const float* __restrict__ u, float* __restrict__ up,
                       float* __restrict__ am) {
    __shared__ float s0[256];
    __shared__ float s1[256];
    const int bc = blockIdx.x;
    const float* p = u + (size_t)bc * HWSZ;
    float s = 0.0f, a = 0.0f;
    for (int i = threadIdx.x; i < HWSZ; i += 256) {
        float v = p[i];
        s += v;
        a += fabsf(v);
    }
    s0[threadIdx.x] = s;
    s1[threadIdx.x] = a;
    __syncthreads();
    for (int off = 128; off > 0; off >>= 1) {
        if (threadIdx.x < off) {
            s0[threadIdx.x] += s0[threadIdx.x + off];
            s1[threadIdx.x] += s1[threadIdx.x + off];
        }
        __syncthreads();
    }
    if (threadIdx.x == 0) {
        up[bc] = s0[0] * (1.0f / HWSZ);
        am[bc] = s1[0] * (1.0f / HWSZ);
    }
}

// ---------------------------------------------------------------------------
// gated O(1) memory update + hist row (tiny: 1 block, 128 threads)
// ---------------------------------------------------------------------------
__global__ void pde_gate_kernel(const float* __restrict__ up,
                                const float* __restrict__ am,
                                const float* __restrict__ Wf,
                                const float* __restrict__ bf,
                                const float* __restrict__ Wi,
                                const float* __restrict__ bi,
                                const float* __restrict__ Wc,
                                const float* __restrict__ bcv,
                                float* __restrict__ h,
                                float* __restrict__ hist_row) {
    const int c = threadIdx.x;  // 0..127
    float hs = 0.0f;
    #pragma unroll
    for (int b = 0; b < BDIM; ++b) hs += am[b * CCH + c];
    hist_row[c] = hs * (1.0f / BDIM);

    float af[BDIM], ai[BDIM], ac[BDIM];
    #pragma unroll
    for (int b = 0; b < BDIM; ++b) { af[b] = 0.f; ai[b] = 0.f; ac[b] = 0.f; }
    for (int k = 0; k < CCH; ++k) {
        float wf = Wf[k * CCH + c];
        float wi = Wi[k * CCH + c];
        float wc = Wc[k * CCH + c];
        #pragma unroll
        for (int b = 0; b < BDIM; ++b) {
            float uv = up[b * CCH + k];
            af[b] = fmaf(uv, wf, af[b]);
            ai[b] = fmaf(uv, wi, ai[b]);
            ac[b] = fmaf(uv, wc, ac[b]);
        }
    }
    #pragma unroll
    for (int b = 0; b < BDIM; ++b) {
        float fg = sigmoidf_(af[b] + bf[c]);
        float ig = sigmoidf_(ai[b] + bi[c]);
        float cg = tanhf(ac[b] + bcv[c]);
        h[b * CCH + c] = h[b * CCH + c] * fg + ig * cg;
    }
}

// ---------------------------------------------------------------------------
// Fused step: stencil + WMMA MLP + memory add + Euler update (+ RMS norm).
// One workgroup = 64 cells (half a spatial row) x all 128 channels.
// LDS: Xh f16[64][136] (34,816? no: 17,408 B) | Hh f16[64][264] (33,792 B,
// aliased as S f32[64][132]) | rinv f32[64]. Total 51,456 B.
// ---------------------------------------------------------------------------
#define XH_STRIDE 136
#define HH_STRIDE 264
#define S_STRIDE  132
#define XH_BYTES  (TCELL * XH_STRIDE * 2)          // 17408
#define HH_BYTES  (TCELL * HH_STRIDE * 2)          // 33792

__global__ __launch_bounds__(256)
void pde_step_kernel(const float* __restrict__ usrc, float* __restrict__ udst,
                     const f16* __restrict__ W1t, const f16* __restrict__ W2t,
                     const float* __restrict__ b1, const float* __restrict__ b2,
                     const float* __restrict__ diff_logits,
                     const float* __restrict__ hgate,
                     const float* __restrict__ scale,
                     const float* __restrict__ log_dt,
                     const float* __restrict__ alpha_mem, int do_norm) {
    __shared__ __align__(16) char smem[XH_BYTES + HH_BYTES + TCELL * 4];
    f16*   Xh   = (f16*)smem;                          // [64][136]
    f16*   Hh   = (f16*)(smem + XH_BYTES);             // [64][264]
    float* S    = (float*)(smem + XH_BYTES);           // alias  [64][132]
    float* rinv = (float*)(smem + XH_BYTES + HH_BYTES);

    const int tid  = threadIdx.x;
    const int lane = tid & 31;
    const int wave = tid >> 5;

    const int wstart = (blockIdx.x & 1) * TCELL;
    const int hrow   = (blockIdx.x >> 1) % HDIM;
    const int b      = blockIdx.x / (2 * HDIM);

    const float dt   = fminf(fmaxf(__expf(log_dt[0]), 0.01f), 0.3f);
    const float amem = alpha_mem[0];
    const float* uplane = usrc + (size_t)b * CCH * HWSZ;

    // ---- stage X tile (64 cells x 128 ch) as f16 --------------------------
    for (int idx = tid; idx < CCH * TCELL; idx += 256) {
        int c = idx >> 6, w = idx & 63;
        Xh[w * XH_STRIDE + c] =
            (f16)uplane[(size_t)c * HWSZ + hrow * WDIM + wstart + w];
    }
    __syncthreads();

    // ---- GEMM1: (64 x 128) @ (128 x 256) ----------------------------------
    // wave owns N columns [wave*32, wave*32+32): 4 M-tiles x 2 N-tiles.
    {
        const int Nb = wave * 32;
        v8f acc[4][2];
        #pragma unroll
        for (int i = 0; i < 4; ++i)
            #pragma unroll
            for (int j = 0; j < 2; ++j)
                acc[i][j] = (v8f){0, 0, 0, 0, 0, 0, 0, 0};

        #pragma unroll
        for (int kk = 0; kk < CCH / 32; ++kk) {
            const int kb = kk * 32;
            v16h a0 = load_a_frag(Xh, XH_STRIDE, 0,  kb, lane);
            v16h a1 = load_a_frag(Xh, XH_STRIDE, 16, kb, lane);
            v16h a2 = load_a_frag(Xh, XH_STRIDE, 32, kb, lane);
            v16h a3 = load_a_frag(Xh, XH_STRIDE, 48, kb, lane);
            v16h bb0 = load_b_frag(W1t, CCH, Nb,      kb, lane);
            v16h bb1 = load_b_frag(W1t, CCH, Nb + 16, kb, lane);
            acc[0][0] = __builtin_amdgcn_wmma_f32_16x16x32_f16(false, a0, false, bb0, (short)0, acc[0][0], false, false);
            acc[0][1] = __builtin_amdgcn_wmma_f32_16x16x32_f16(false, a0, false, bb1, (short)0, acc[0][1], false, false);
            acc[1][0] = __builtin_amdgcn_wmma_f32_16x16x32_f16(false, a1, false, bb0, (short)0, acc[1][0], false, false);
            acc[1][1] = __builtin_amdgcn_wmma_f32_16x16x32_f16(false, a1, false, bb1, (short)0, acc[1][1], false, false);
            acc[2][0] = __builtin_amdgcn_wmma_f32_16x16x32_f16(false, a2, false, bb0, (short)0, acc[2][0], false, false);
            acc[2][1] = __builtin_amdgcn_wmma_f32_16x16x32_f16(false, a2, false, bb1, (short)0, acc[2][1], false, false);
            acc[3][0] = __builtin_amdgcn_wmma_f32_16x16x32_f16(false, a3, false, bb0, (short)0, acc[3][0], false, false);
            acc[3][1] = __builtin_amdgcn_wmma_f32_16x16x32_f16(false, a3, false, bb1, (short)0, acc[3][1], false, false);
        }

        // epilogue: +b1, exact GELU, f16 -> Hh
        #pragma unroll
        for (int i = 0; i < 4; ++i) {
            #pragma unroll
            for (int j = 0; j < 2; ++j) {
                const int n = Nb + j * 16 + (lane & 15);
                const float bn = b1[n];
                #pragma unroll
                for (int r = 0; r < 8; ++r) {
                    const int m = i * 16 + r + 8 * (lane >> 4);
                    float v = acc[i][j][r] + bn;
                    float g = 0.5f * v * (1.0f + erff(v * 0.70710678118654752f));
                    Hh[m * HH_STRIDE + n] = (f16)g;
                }
            }
        }
    }
    __syncthreads();

    // ---- GEMM2: (64 x 256) @ (256 x 128) ----------------------------------
    // wave owns N columns [wave*16, wave*16+16): 4 M-tiles x 1 N-tile.
    v8f acc2[4];
    {
        const int Nb = wave * 16;
        #pragma unroll
        for (int i = 0; i < 4; ++i) acc2[i] = (v8f){0, 0, 0, 0, 0, 0, 0, 0};

        #pragma unroll
        for (int kk = 0; kk < HID / 32; ++kk) {
            const int kb = kk * 32;
            v16h a0 = load_a_frag(Hh, HH_STRIDE, 0,  kb, lane);
            v16h a1 = load_a_frag(Hh, HH_STRIDE, 16, kb, lane);
            v16h a2 = load_a_frag(Hh, HH_STRIDE, 32, kb, lane);
            v16h a3 = load_a_frag(Hh, HH_STRIDE, 48, kb, lane);
            v16h bb = load_b_frag(W2t, HID, Nb, kb, lane);
            acc2[0] = __builtin_amdgcn_wmma_f32_16x16x32_f16(false, a0, false, bb, (short)0, acc2[0], false, false);
            acc2[1] = __builtin_amdgcn_wmma_f32_16x16x32_f16(false, a1, false, bb, (short)0, acc2[1], false, false);
            acc2[2] = __builtin_amdgcn_wmma_f32_16x16x32_f16(false, a2, false, bb, (short)0, acc2[2], false, false);
            acc2[3] = __builtin_amdgcn_wmma_f32_16x16x32_f16(false, a3, false, bb, (short)0, acc2[3], false, false);
        }
    }
    __syncthreads();   // Hh reads done everywhere before aliasing as S

    // ---- epilogue: Laplacian + react + memory, Euler update ---------------
    {
        const int cg = (wave * 16) + (lane & 15);     // channel, fixed per thread
        const float cf0 = 0.25f * sigmoidf_(diff_logits[0 * CCH + cg]);
        const float cf1 = 0.25f * sigmoidf_(diff_logits[1 * CCH + cg]);
        const float cf2 = 0.25f * sigmoidf_(diff_logits[2 * CCH + cg]);
        const float hb  = amem * hgate[b * CCH + cg];
        const float b2c = b2[cg];
        const float* up2 = uplane + (size_t)cg * HWSZ;
        const int d_[3] = {1, 2, 4};
        const float cf_[3] = {cf0, cf1, cf2};

        #pragma unroll
        for (int i = 0; i < 4; ++i) {
            #pragma unroll
            for (int r = 0; r < 8; ++r) {
                const int wl = i * 16 + r + 8 * (lane >> 4);  // local cell
                const int wg = wstart + wl;
                const float center = up2[hrow * WDIM + wg];
                float diffv = 0.0f;
                #pragma unroll
                for (int di = 0; di < 3; ++di) {
                    const int d = d_[di];
                    const int wlft = reflect_(wg - d, WDIM);
                    const int wrgt = reflect_(wg + d, WDIM);
                    const int hup  = reflect_(hrow - d, HDIM);
                    const int hdn  = reflect_(hrow + d, HDIM);
                    float lap = up2[hrow * WDIM + wlft] + up2[hrow * WDIM + wrgt] +
                                up2[hup * WDIM + wg] + up2[hdn * WDIM + wg] -
                                4.0f * center;
                    diffv = fmaf(cf_[di], lap, diffv);
                }
                const float react = acc2[i][r] + b2c;
                const float du = diffv + react + hb;
                S[wl * S_STRIDE + cg] = fmaf(dt, du, center);
            }
        }
    }
    __syncthreads();

    // ---- optional RMS norm + coalesced store ------------------------------
    if (do_norm) {
        if (tid < TCELL) {
            float ss = 0.0f;
            for (int c = 0; c < CCH; ++c) {
                float v = S[tid * S_STRIDE + c];
                ss = fmaf(v, v, ss);
            }
            float rms = sqrtf(ss) * 0.08838834764831845f;  // * 1/sqrt(128)
            rinv[tid] = 1.0f / (rms + 1e-6f);
        }
        __syncthreads();
        float* dplane = udst + (size_t)b * CCH * HWSZ;
        for (int idx = tid; idx < CCH * TCELL; idx += 256) {
            int c = idx >> 6, w = idx & 63;
            dplane[(size_t)c * HWSZ + hrow * WDIM + wstart + w] =
                S[w * S_STRIDE + c] * rinv[w] * scale[c];
        }
    } else {
        float* dplane = udst + (size_t)b * CCH * HWSZ;
        for (int idx = tid; idx < CCH * TCELL; idx += 256) {
            int c = idx >> 6, w = idx & 63;
            dplane[(size_t)c * HWSZ + hrow * WDIM + wstart + w] =
                S[w * S_STRIDE + c];
        }
    }
}

// ---------------------------------------------------------------------------
// finalize: channel_velocity from channel_history, and dt scalar
// ---------------------------------------------------------------------------
__global__ void pde_finalize_kernel(const float* __restrict__ hist,
                                    float* __restrict__ vel,
                                    float* __restrict__ dt_out,
                                    const float* __restrict__ log_dt) {
    const int c = threadIdx.x;
    float s = 0.0f;
    #pragma unroll
    for (int t = 0; t < 11; ++t)
        s += fabsf(hist[(t + 1) * CCH + c] - hist[t * CCH + c]);
    vel[c] = s * (1.0f / 11.0f);
    if (c == 0)
        dt_out[0] = fminf(fmaxf(__expf(log_dt[0]), 0.01f), 0.3f);
}

// ---------------------------------------------------------------------------
extern "C" void kernel_launch(void* const* d_in, const int* in_sizes, int n_in,
                              void* d_out, int out_size, void* d_ws,
                              size_t ws_size, hipStream_t stream) {
    (void)in_sizes; (void)n_in; (void)out_size; (void)ws_size;
    const float* u_in   = (const float*)d_in[0];
    const float* dlg    = (const float*)d_in[1];
    const float* W1     = (const float*)d_in[2];
    const float* b1     = (const float*)d_in[3];
    const float* W2     = (const float*)d_in[4];
    const float* b2     = (const float*)d_in[5];
    const float* Wf     = (const float*)d_in[6];
    const float* bf     = (const float*)d_in[7];
    const float* Wi     = (const float*)d_in[8];
    const float* bi     = (const float*)d_in[9];
    const float* Wc     = (const float*)d_in[10];
    const float* bcv    = (const float*)d_in[11];
    const float* scale  = (const float*)d_in[12];
    const float* log_dt = (const float*)d_in[13];
    const float* amem   = (const float*)d_in[14];

    float* out       = (float*)d_out;
    float* out_u     = out;                           // B*C*H*W = 16777216
    float* out_vel   = out + (size_t)BDIM * CCH * HWSZ;
    float* out_hist  = out_vel + CCH;                 // 12*C
    float* out_dt    = out_hist + 12 * CCH;

    char* ws = (char*)d_ws;
    f16*   W1t  = (f16*)ws;                           // 64 KB
    f16*   W2t  = (f16*)(ws + 65536);                 // 64 KB
    float* h    = (float*)(ws + 131072);              // B*C
    float* up   = (float*)(ws + 135168);              // B*C
    float* am   = (float*)(ws + 139264);              // B*C
    float* ubuf = (float*)(ws + 143360);              // 64 MB ping buffer

    pde_setup_kernel<<<260, 256, 0, stream>>>(W1, W2, W1t, W2t, h);

    const float* src = u_in;
    for (int t = 0; t < 12; ++t) {
        float* dst = (t & 1) ? out_u : ubuf;          // t=11 lands in d_out
        pde_reduce_kernel<<<BDIM * CCH, 256, 0, stream>>>(src, up, am);
        pde_gate_kernel<<<1, CCH, 0, stream>>>(up, am, Wf, bf, Wi, bi, Wc, bcv,
                                               h, out_hist + t * CCH);
        pde_step_kernel<<<BDIM * HDIM * 2, 256, 0, stream>>>(
            src, dst, W1t, W2t, b1, b2, dlg, h, scale, log_dt, amem,
            ((t + 1) & 1) == 0);
        src = dst;
    }
    pde_finalize_kernel<<<1, CCH, 0, stream>>>(out_hist, out_vel, out_dt,
                                               log_dt);
}
